// LSTM_Turnoverv2_64690797412522
// MI455X (gfx1250) — matrix operand order repaired and checked by reading, not compiled, not run on hardware
//
#include <hip/hip_runtime.h>

// ---------------- problem constants (from reference) ----------------
#define TT    256          // input timesteps
#define FUTN  32           // autoregressive future steps
#define TOTN  (TT + FUTN)  // 288
#define HDN   48           // hidden size
#define NGATE 192          // 4*H
#define NTILE 12           // 192 / 16
#define K0CH  2            // cell0: K=64 -> 2 chunks of 32
#define K1CH  3            // cell1: K=96 -> 3 chunks of 32
#define PITCH0 64          // staging row pitch (halfs) for cell0 A
#define PITCH1 96          // staging row pitch (halfs) for cell1 A
#define WAVES  2           // waves per block (16 batch rows per wave)

typedef __attribute__((ext_vector_type(16))) _Float16 v16h;
typedef __attribute__((ext_vector_type(8)))  float    v8f;

__device__ __forceinline__ float t_tanh(float x) {
#if __has_builtin(__builtin_amdgcn_tanhf)
  return __builtin_amdgcn_tanhf(x);
#else
  return tanhf(x);
#endif
}
__device__ __forceinline__ float t_sigm(float x) {
  return 0.5f * t_tanh(0.5f * x) + 0.5f;
}

// Build a 16x32 f16 A-fragment from an LDS-staged row-major [16][pitch] tile.
// A layout used here (contiguous k, same as packed B): lane m = lane%16, hi = lane/16;
// half e: pair p=e/2 -> K = kc*32 + 16*hi + 2p + (e&1)
__device__ __forceinline__ v16h build_afrag(const _Float16* S, int m, int hi,
                                            int kc, int pitch) {
  union { v16h v; unsigned u[8]; } au;
  const unsigned* row = (const unsigned*)(S + m * pitch);
  int kb = kc * 32 + 16 * hi;
#pragma unroll
  for (int p = 0; p < 8; ++p) {
    int kk = kb + 2 * p;
    au.u[p] = row[kk >> 1];
  }
  return au.v;
}

// ---------------- kernel 1: input encoders -> add_enc[B][4] ----------------
__global__ void enc_kernel(int B,
  const float* __restrict__ dep, const float* __restrict__ idr,
  const float* __restrict__ zod, const float* __restrict__ addin,
  const float* __restrict__ idr_w1, const float* __restrict__ idr_b1,
  const float* __restrict__ idr_w2, const float* __restrict__ idr_b2,
  const float* __restrict__ zod_w1, const float* __restrict__ zod_b1,
  const float* __restrict__ zod_w2, const float* __restrict__ zod_b2,
  const float* __restrict__ dep_w1, const float* __restrict__ dep_b1,
  const float* __restrict__ dep_w2, const float* __restrict__ dep_b2,
  const float* __restrict__ inp_w1, const float* __restrict__ inp_b1,
  const float* __restrict__ inp_w2, const float* __restrict__ inp_b2,
  float* __restrict__ add_enc)
{
  int b = blockIdx.x * blockDim.x + threadIdx.x;
  if (b >= B) return;
  float v15[15];
  v15[0] = addin[b*3+0]; v15[1] = addin[b*3+1]; v15[2] = addin[b*3+2];
  float t1[16], t2[16];
  // dep: 4 -> 8 -> 4 (tanh, tanh)
  for (int q = 0; q < 8; ++q) { float z = dep_b1[q];
    for (int j = 0; j < 4; ++j) z += dep[b*4+j] * dep_w1[q*4+j];
    t1[q] = t_tanh(z); }
  for (int q = 0; q < 4; ++q) { float z = dep_b2[q];
    for (int j = 0; j < 8; ++j) z += t1[j] * dep_w2[q*8+j];
    v15[3+q] = t_tanh(z); }
  // zod: 8 -> 6 -> 4
  for (int q = 0; q < 6; ++q) { float z = zod_b1[q];
    for (int j = 0; j < 8; ++j) z += zod[b*8+j] * zod_w1[q*8+j];
    t1[q] = t_tanh(z); }
  for (int q = 0; q < 4; ++q) { float z = zod_b2[q];
    for (int j = 0; j < 6; ++j) z += t1[j] * zod_w2[q*6+j];
    v15[7+q] = t_tanh(z); }
  // idr: 29 -> 16 -> 4
  for (int q = 0; q < 16; ++q) { float z = idr_b1[q];
    for (int j = 0; j < 29; ++j) z += idr[b*29+j] * idr_w1[q*29+j];
    t1[q] = t_tanh(z); }
  for (int q = 0; q < 4; ++q) { float z = idr_b2[q];
    for (int j = 0; j < 16; ++j) z += t1[j] * idr_w2[q*16+j];
    v15[11+q] = t_tanh(z); }
  // input encoder: 15 -> 16 -> 4
  for (int q = 0; q < 16; ++q) { float z = inp_b1[q];
    for (int j = 0; j < 15; ++j) z += v15[j] * inp_w1[q*15+j];
    t2[q] = t_tanh(z); }
  for (int q = 0; q < 4; ++q) { float z = inp_b2[q];
    for (int j = 0; j < 16; ++j) z += t2[j] * inp_w2[q*16+j];
    add_enc[b*4+q] = t_tanh(z); }
}

// ---------------- kernel 2: pack LSTM weights into WMMA B-fragments ----------
// B-fragment (f16, 32x16): lane n = lane%16, hi = lane/16,
// half e: K = kc*32 + 16*hi + e.  Stored lane-major: frag[f][lane][16 halfs].
// cell0 Wcat0[K=64][N=192]: K<5 -> w_ih0[n][K]; 5<=K<53 -> w_hh0[n][K-5]; else 0
// cell1 Wcat1[K=96][N=192]: K<48 -> w_ih1[n][K]; else w_hh1[n][K-48]
__global__ void pack_kernel(
  const float* __restrict__ w_ih0, const float* __restrict__ w_hh0,
  const float* __restrict__ b_ih0, const float* __restrict__ b_hh0,
  const float* __restrict__ w_ih1, const float* __restrict__ w_hh1,
  const float* __restrict__ b_ih1, const float* __restrict__ b_hh1,
  _Float16* __restrict__ W0, _Float16* __restrict__ W1,
  float* __restrict__ bias0, float* __restrict__ bias1)
{
  int tid = blockIdx.x * blockDim.x + threadIdx.x;
  if (tid < NGATE) {
    bias0[tid] = b_ih0[tid] + b_hh0[tid];
    bias1[tid] = b_ih1[tid] + b_hh1[tid];
  }
  if (tid < NTILE * K0CH * 32) {
    int f = tid >> 5, lane = tid & 31;
    int kc = f % K0CH, tile = f / K0CH;
    int nn = tile * 16 + (lane & 15), hi = lane >> 4;
    for (int e = 0; e < 16; ++e) {
      int kk = kc * 32 + 16 * hi + e;
      float v = 0.f;
      if (kk < 5)        v = w_ih0[nn * 5 + kk];
      else if (kk < 53)  v = w_hh0[nn * HDN + (kk - 5)];
      W0[f * 512 + lane * 16 + e] = (_Float16)v;
    }
  }
  if (tid < NTILE * K1CH * 32) {
    int f = tid >> 5, lane = tid & 31;
    int kc = f % K1CH, tile = f / K1CH;
    int nn = tile * 16 + (lane & 15), hi = lane >> 4;
    for (int e = 0; e < 16; ++e) {
      int kk = kc * 32 + 16 * hi + e;
      float v = (kk < HDN) ? w_ih1[nn * HDN + kk]
                           : w_hh1[nn * HDN + (kk - HDN)];
      W1[f * 512 + lane * 16 + e] = (_Float16)v;
    }
  }
}

// ---------------- kernel 3: the recurrent LSTM, WMMA per step ----------------
#define WMMA16(A, Bf, C) \
  __builtin_amdgcn_wmma_f32_16x16x32_f16(false, (A), false, (Bf), (short)0, (C), false, false)

// dynamic LDS layout (halfs / floats), see kernel_launch for total size
#define SW0_H   (NTILE * K0CH * 512)            // 12288 halfs
#define SW1_H   (NTILE * K1CH * 512)            // 18432 halfs
#define ST0_H   (WAVES * 16 * PITCH0)           // 2048 halfs
#define ST1_H   (WAVES * 16 * PITCH1)           // 3072 halfs
#define H1F_F   (WAVES * 16 * HDN)              // 1536 floats
#define SMALL_F (HDN + 80 + 16 + 16)            // 160 floats
#define SMEM_BYTES ((SW0_H + SW1_H + ST0_H + ST1_H) * 2 + (H1F_F + SMALL_F) * 4)

__global__ __launch_bounds__(WAVES * 32) void lstm_main(int B,
  const float* __restrict__ x, const float* __restrict__ add_enc,
  const _Float16* __restrict__ W0g, const _Float16* __restrict__ W1g,
  const float* __restrict__ bias0, const float* __restrict__ bias1,
  const float* __restrict__ lin_w, const float* __restrict__ lin_b,
  const float* __restrict__ out_w1, const float* __restrict__ out_b1,
  const float* __restrict__ out_w2, const float* __restrict__ out_b2,
  float* __restrict__ out)
{
  extern __shared__ __align__(32) char smem[];
  _Float16* sW0 = (_Float16*)smem;
  _Float16* sW1 = sW0 + SW0_H;
  _Float16* st0 = sW1 + SW1_H;
  _Float16* st1 = st0 + ST0_H;
  float* h1f   = (float*)(st1 + ST1_H);
  float* s_lin = h1f + H1F_F;
  float* s_ow1 = s_lin + HDN;   // 80
  float* s_ob1 = s_ow1 + 80;    // 16
  float* s_ow2 = s_ob1 + 16;    // 16

  const int tid  = threadIdx.x;
  const int w    = tid >> 5;
  const int lane = tid & 31;
  const int n    = lane & 15;
  const int hi   = lane >> 4;
  const int rowbase = (blockIdx.x * WAVES + w) * 16;

  // block init: weights + head params into LDS
  for (int i = tid; i < SW0_H; i += blockDim.x) sW0[i] = W0g[i];
  for (int i = tid; i < SW1_H; i += blockDim.x) sW1[i] = W1g[i];
  if (tid < HDN) s_lin[tid] = lin_w[tid];
  if (tid < 16) { s_ob1[tid] = out_b1[tid]; s_ow2[tid] = out_w2[tid]; }
  for (int i = tid; i < 80; i += blockDim.x) s_ow1[i] = out_w1[i];

  _Float16* S0 = st0 + w * 16 * PITCH0;
  _Float16* S1 = st1 + w * 16 * PITCH1;
  float*    HF = h1f + w * 16 * HDN;
  for (int i = lane; i < 16 * PITCH0; i += 32) S0[i] = (_Float16)0.f;
  for (int i = lane; i < 16 * PITCH1; i += 32) S1[i] = (_Float16)0.f;

  float aef[4] = {0.f, 0.f, 0.f, 0.f};
  if (hi == 0) {  // lanes 0..15 each own batch row rowbase+n
    for (int j = 0; j < 4; ++j) {
      aef[j] = add_enc[(size_t)(rowbase + n) * 4 + j];
      S0[n * PITCH0 + 1 + j] = (_Float16)aef[j];   // constant input cols 1..4
    }
  }
  __syncthreads();

  float bg0[NTILE], bg1[NTILE];
#pragma unroll
  for (int i = 0; i < NTILE; ++i) { bg0[i] = bias0[i*16 + n]; bg1[i] = bias1[i*16 + n]; }
  const float lb = lin_b[0], ob2 = out_b2[0];

  v8f c0[3] = {}; v8f c1[3] = {};
  float yprev = 0.f;

  for (int t = 0; t < TOTN; ++t) {
    // ---- stage input column 0 (x_t in phase 1, y_prev in phase 2) ----
    if (hi == 0) {
      float xin = (t < TT) ? x[(size_t)(rowbase + n) * TT + t] : yprev;
      S0[n * PITCH0] = (_Float16)xin;
    }

    // ---- cell 0: gates = [x|add_enc|h0] (K=64) * Wcat0 + bias ----
    v16h a00 = build_afrag(S0, n, hi, 0, PITCH0);
    v16h a01 = build_afrag(S0, n, hi, 1, PITCH0);
    v8f acc[NTILE];
#pragma unroll
    for (int tt2 = 0; tt2 < NTILE; ++tt2) {
      v8f a;
#pragma unroll
      for (int r = 0; r < 8; ++r) a[r] = bg0[tt2];
      v16h b0 = *(const v16h*)(sW0 + ((tt2 * K0CH + 0) * 512 + lane * 16));
      a = WMMA16(a00, b0, a);
      v16h b1 = *(const v16h*)(sW0 + ((tt2 * K0CH + 1) * 512 + lane * 16));
      a = WMMA16(a01, b1, a);
      acc[tt2] = a;
    }
    // gate order i(0:48) f(48:96) g(96:144) o(144:192) -> tiles [0..2][3..5][6..8][9..11]
#pragma unroll
    for (int tt2 = 0; tt2 < 3; ++tt2) {
      float hn[8];
#pragma unroll
      for (int r = 0; r < 8; ++r) {
        float ig = t_sigm(acc[tt2][r]);
        float fg = t_sigm(acc[3 + tt2][r]);
        float gg = t_tanh(acc[6 + tt2][r]);
        float og = t_sigm(acc[9 + tt2][r]);
        float cn = fg * c0[tt2][r] + ig * gg;
        c0[tt2][r] = cn;
        hn[r] = og * t_tanh(cn);
      }
#pragma unroll
      for (int r = 0; r < 8; ++r) {     // D layout: row m = r + 8*hi, col = tile*16+n
        int m = r + 8 * hi;
        _Float16 hv = (_Float16)hn[r];
        S0[m * PITCH0 + 5 + tt2 * 16 + n] = hv;   // h0 for next step's cell0 A
        S1[m * PITCH1 +     tt2 * 16 + n] = hv;   // h0 for this step's cell1 A
      }
    }

    // ---- cell 1: gates = [h0|h1] (K=96) * Wcat1 + bias ----
    v16h a10 = build_afrag(S1, n, hi, 0, PITCH1);
    v16h a11 = build_afrag(S1, n, hi, 1, PITCH1);
    v16h a12 = build_afrag(S1, n, hi, 2, PITCH1);
#pragma unroll
    for (int tt2 = 0; tt2 < NTILE; ++tt2) {
      v8f a;
#pragma unroll
      for (int r = 0; r < 8; ++r) a[r] = bg1[tt2];
      v16h b0 = *(const v16h*)(sW1 + ((tt2 * K1CH + 0) * 512 + lane * 16));
      a = WMMA16(a10, b0, a);
      v16h b1 = *(const v16h*)(sW1 + ((tt2 * K1CH + 1) * 512 + lane * 16));
      a = WMMA16(a11, b1, a);
      v16h b2 = *(const v16h*)(sW1 + ((tt2 * K1CH + 2) * 512 + lane * 16));
      a = WMMA16(a12, b2, a);
      acc[tt2] = a;
    }
#pragma unroll
    for (int tt2 = 0; tt2 < 3; ++tt2) {
#pragma unroll
      for (int r = 0; r < 8; ++r) {
        float ig = t_sigm(acc[tt2][r]);
        float fg = t_sigm(acc[3 + tt2][r]);
        float gg = t_tanh(acc[6 + tt2][r]);
        float og = t_sigm(acc[9 + tt2][r]);
        float cn = fg * c1[tt2][r] + ig * gg;
        c1[tt2][r] = cn;
        float hv = og * t_tanh(cn);
        int m = r + 8 * hi;
        S1[m * PITCH1 + HDN + tt2 * 16 + n] = (_Float16)hv;  // h1 for next step
        HF[m * HDN + tt2 * 16 + n] = hv;                      // f32 for head
      }
    }

    // ---- output head (lanes 0..15, one batch row each) ----
    if (hi == 0) {
      float o = lb;
#pragma unroll
      for (int j = 0; j < HDN; ++j) o += HF[n * HDN + j] * s_lin[j];
      float y = ob2;
#pragma unroll
      for (int q = 0; q < 16; ++q) {
        float z = s_ob1[q] + o * s_ow1[q*5]
                + aef[0] * s_ow1[q*5+1] + aef[1] * s_ow1[q*5+2]
                + aef[2] * s_ow1[q*5+3] + aef[3] * s_ow1[q*5+4];
        y += t_tanh(z) * s_ow2[q];
      }
      out[(size_t)(rowbase + n) * TOTN + t] = y;
      yprev = y;
    }
  }
}

// ---------------- host launcher ----------------
extern "C" void kernel_launch(void* const* d_in, const int* in_sizes, int n_in,
                              void* d_out, int out_size, void* d_ws, size_t ws_size,
                              hipStream_t stream) {
  const float* x      = (const float*)d_in[0];
  const float* dep    = (const float*)d_in[1];
  const float* idr    = (const float*)d_in[2];
  const float* zod    = (const float*)d_in[3];
  const float* addin  = (const float*)d_in[4];
  const float* idr_w1 = (const float*)d_in[5];
  const float* idr_b1 = (const float*)d_in[6];
  const float* idr_w2 = (const float*)d_in[7];
  const float* idr_b2 = (const float*)d_in[8];
  const float* zod_w1 = (const float*)d_in[9];
  const float* zod_b1 = (const float*)d_in[10];
  const float* zod_w2 = (const float*)d_in[11];
  const float* zod_b2 = (const float*)d_in[12];
  const float* dep_w1 = (const float*)d_in[13];
  const float* dep_b1 = (const float*)d_in[14];
  const float* dep_w2 = (const float*)d_in[15];
  const float* dep_b2 = (const float*)d_in[16];
  const float* inp_w1 = (const float*)d_in[17];
  const float* inp_b1 = (const float*)d_in[18];
  const float* inp_w2 = (const float*)d_in[19];
  const float* inp_b2 = (const float*)d_in[20];
  const float* out_w1 = (const float*)d_in[21];
  const float* out_b1 = (const float*)d_in[22];
  const float* out_w2 = (const float*)d_in[23];
  const float* out_b2 = (const float*)d_in[24];
  const float* lin_w  = (const float*)d_in[25];
  const float* lin_b  = (const float*)d_in[26];
  const float* w_ih0  = (const float*)d_in[27];
  const float* w_hh0  = (const float*)d_in[28];
  const float* b_ih0  = (const float*)d_in[29];
  const float* b_hh0  = (const float*)d_in[30];
  const float* w_ih1  = (const float*)d_in[31];
  const float* w_hh1  = (const float*)d_in[32];
  const float* b_ih1  = (const float*)d_in[33];
  const float* b_hh1  = (const float*)d_in[34];
  (void)n_in; (void)out_size; (void)ws_size;

  const int B = in_sizes[0] / TT;   // 4096

  // workspace carve-out
  char* wsb = (char*)d_ws;
  size_t off = 0;
  float* add_enc = (float*)(wsb + off); off += (size_t)B * 4 * sizeof(float);
  float* bias0   = (float*)(wsb + off); off += NGATE * sizeof(float);
  float* bias1   = (float*)(wsb + off); off += NGATE * sizeof(float);
  off = (off + 31) & ~(size_t)31;
  _Float16* W0p  = (_Float16*)(wsb + off); off += (size_t)NTILE * K0CH * 512 * 2;
  _Float16* W1p  = (_Float16*)(wsb + off); off += (size_t)NTILE * K1CH * 512 * 2;

  enc_kernel<<<(B + 127) / 128, 128, 0, stream>>>(B,
      dep, idr, zod, addin,
      idr_w1, idr_b1, idr_w2, idr_b2,
      zod_w1, zod_b1, zod_w2, zod_b2,
      dep_w1, dep_b1, dep_w2, dep_b2,
      inp_w1, inp_b1, inp_w2, inp_b2,
      add_enc);

  pack_kernel<<<5, 256, 0, stream>>>(
      w_ih0, w_hh0, b_ih0, b_hh0,
      w_ih1, w_hh1, b_ih1, b_hh1,
      W0p, W1p, bias0, bias1);

  lstm_main<<<B / (16 * WAVES), WAVES * 32, SMEM_BYTES, stream>>>(B,
      x, add_enc, W0p, W1p, bias0, bias1,
      lin_w, lin_b, out_w1, out_b1, out_w2, out_b2,
      (float*)d_out);
}
